// SelfAttention_7292854468827
// MI455X (gfx1250) — matrix operand order repaired and checked by reading, not compiled, not run on hardware
//
#include <hip/hip_runtime.h>

#define B_  4
#define S_  2048
#define D_  2048
#define H_  16
#define HD_ 128

typedef __attribute__((ext_vector_type(16))) __bf16 bf16x16;
typedef __attribute__((ext_vector_type(8)))  __bf16 bf16x8;
typedef __attribute__((ext_vector_type(8)))  float   v8f;
typedef __attribute__((ext_vector_type(4)))  float   f32x4;
typedef __attribute__((ext_vector_type(4)))  unsigned u32x4;
typedef __attribute__((ext_vector_type(8)))  int      i32x8;
typedef __attribute__((ext_vector_type(4)))  int      i32x4;

#if defined(__HIP_DEVICE_COMPILE__) && __has_builtin(__builtin_amdgcn_tensor_load_to_lds)
#define HAVE_TDM 1
#else
#define HAVE_TDM 0
#endif

// ---- WMMA helpers -----------------------------------------------------------

__device__ __forceinline__ v8f wmma_bf16(bf16x16 a, bf16x16 b, v8f c) {
  // D = A(16x32 bf16) * B(32x16 bf16) + C(16x16 f32)
  return __builtin_amdgcn_wmma_f32_16x16x32_bf16(
      /*neg_a=*/false, a, /*neg_b=*/false, b,
      /*c_mod=*/(short)0, c, /*reuse_a=*/false, /*reuse_b=*/false);
}

// A-matrix 16x32 bf16 fragment: lane (half,lr) holds row M=lr,
// K = {half*8 .. half*8+7} U {16+half*8 .. 16+half*8+7}
__device__ __forceinline__ bf16x16 a_frag(const __bf16* row, int half) {
  bf16x8 lo = *(const bf16x8*)(row + half * 8);
  bf16x8 hi = *(const bf16x8*)(row + 16 + half * 8);
  bf16x16 r;
#pragma unroll
  for (int i = 0; i < 8; ++i) { r[i] = lo[i]; r[i + 8] = hi[i]; }
  return r;
}

// B-matrix 32x16 bf16 fragment: lane (half,lr) holds col N=lr,
// K = {half*8 .. half*8+7} U {16+half*8 .. 16+half*8+7}.  `col` points at that lane's
// 32 K-values (B stored N-major / transposed in LDS).
__device__ __forceinline__ bf16x16 b_frag(const __bf16* col, int half) {
  bf16x8 lo = *(const bf16x8*)(col + half * 8);
  bf16x8 hi = *(const bf16x8*)(col + 16 + half * 8);
  bf16x16 r;
#pragma unroll
  for (int i = 0; i < 8; ++i) { r[i] = lo[i]; r[i + 8] = hi[i]; }
  return r;
}

// ---- Kernel 1: QKV projection GEMM (fp32 in -> bf16 out) -------------------
// C[M=8192, N=2048] = X[M,K=2048] * W[K,N], one of wq/wk/wv per blockIdx.z.

__global__ __launch_bounds__(256)
void qkv_gemm_kernel(const float* __restrict__ x,
                     const float* __restrict__ wq,
                     const float* __restrict__ wk,
                     const float* __restrict__ wv,
                     __bf16* __restrict__ qws,
                     __bf16* __restrict__ kws,
                     __bf16* __restrict__ vws) {
  constexpr int K = D_, N = H_ * HD_;
  const int which = blockIdx.z;
  const float* wmat = (which == 0) ? wq : (which == 1) ? wk : wv;
  __bf16* outp     = (which == 0) ? qws : (which == 1) ? kws : vws;

  __shared__ __bf16 lda[128][40];   // A tile, row-major [m][k], padded
  __shared__ __bf16 ldb[128][40];   // B tile, transposed [n][k], padded

  const int tid  = threadIdx.x;
  const int lane = tid & 31, w = tid >> 5;
  const int half = lane >> 4, lr = lane & 15;
  const int wm = (w & 1) * 64;      // wave row offset in block tile
  const int wn = (w >> 1) * 32;     // wave col offset in block tile
  const int m0 = blockIdx.y * 128;
  const int n0 = blockIdx.x * 128;

  v8f zero = {0.f, 0.f, 0.f, 0.f, 0.f, 0.f, 0.f, 0.f};
  v8f acc[4][2];
#pragma unroll
  for (int i = 0; i < 4; ++i)
#pragma unroll
    for (int j = 0; j < 2; ++j) acc[i][j] = zero;

  const int arow = tid & 127, ahalf = tid >> 7;  // A staging: 128 rows x 2 k-halves
  const int brow = tid & 31,  bpart = tid >> 5;  // B staging: 32 k-rows x 8 n-chunks

  for (int kb = 0; kb < K; kb += 32) {
    // Stage A 128x32 (fp32 -> bf16, contiguous)
    {
      const float* src = x + (size_t)(m0 + arow) * K + kb + ahalf * 16;
      f32x4 v0 = *(const f32x4*)(src + 0);
      f32x4 v1 = *(const f32x4*)(src + 4);
      f32x4 v2 = *(const f32x4*)(src + 8);
      f32x4 v3 = *(const f32x4*)(src + 12);
      bf16x8 t0, t1;
#pragma unroll
      for (int e = 0; e < 4; ++e) {
        t0[e] = (__bf16)v0[e]; t0[e + 4] = (__bf16)v1[e];
        t1[e] = (__bf16)v2[e]; t1[e + 4] = (__bf16)v3[e];
      }
      *(bf16x8*)(&lda[arow][ahalf * 16 + 0]) = t0;
      *(bf16x8*)(&lda[arow][ahalf * 16 + 8]) = t1;
    }
    // Stage B 32x128 transposed into [n][k] (fp32 -> bf16)
    {
      const float* src = wmat + (size_t)(kb + brow) * N + n0 + bpart * 16;
      f32x4 v0 = *(const f32x4*)(src + 0);
      f32x4 v1 = *(const f32x4*)(src + 4);
      f32x4 v2 = *(const f32x4*)(src + 8);
      f32x4 v3 = *(const f32x4*)(src + 12);
#pragma unroll
      for (int e = 0; e < 4; ++e) {
        ldb[bpart * 16 + e +  0][brow] = (__bf16)v0[e];
        ldb[bpart * 16 + e +  4][brow] = (__bf16)v1[e];
        ldb[bpart * 16 + e +  8][brow] = (__bf16)v2[e];
        ldb[bpart * 16 + e + 12][brow] = (__bf16)v3[e];
      }
    }
    if (kb + 32 < K) {
      __builtin_prefetch(x + (size_t)(m0 + arow) * K + kb + 32 + ahalf * 16, 0, 1);
      __builtin_prefetch(wmat + (size_t)(kb + 32 + brow) * N + n0 + bpart * 16, 0, 1);
    }
    __syncthreads();

    bf16x16 af[4], bfr[2];
#pragma unroll
    for (int i = 0; i < 4; ++i) af[i] = a_frag(&lda[wm + i * 16 + lr][0], half);
#pragma unroll
    for (int j = 0; j < 2; ++j) bfr[j] = b_frag(&ldb[wn + j * 16 + lr][0], half);
#pragma unroll
    for (int i = 0; i < 4; ++i)
#pragma unroll
      for (int j = 0; j < 2; ++j) acc[i][j] = wmma_bf16(af[i], bfr[j], acc[i][j]);
    __syncthreads();
  }

  // Epilogue: C layout lane(half,lr) VGPR r -> (M = r+8*half, N = lr)
#pragma unroll
  for (int i = 0; i < 4; ++i) {
    const int rowb = m0 + wm + i * 16 + 8 * half;
#pragma unroll
    for (int j = 0; j < 2; ++j) {
      const int col = n0 + wn + j * 16 + lr;
#pragma unroll
      for (int r = 0; r < 8; ++r)
        outp[(size_t)(rowb + r) * N + col] = (__bf16)acc[i][j][r];
    }
  }
}

// ---- Kernel 2: RoPE on q,k (in-place, bf16) --------------------------------

__global__ __launch_bounds__(256)
void rope_kernel(__bf16* __restrict__ q, __bf16* __restrict__ k) {
  const int bs = blockIdx.x;               // b*S + s
  const int s  = bs & (S_ - 1);
  const int hh = threadIdx.x >> 6;         // 0..3
  const int d  = threadIdx.x & 63;         // 0..63 (pairs with d+64)
  const int h  = blockIdx.y * 4 + hh;
  const size_t base = ((size_t)bs * H_ + h) * HD_;

  const float kLn1e4_64 = 0.14391156831212787f;  // ln(10000)/64
  const float inv_freq = __expf(-(float)d * kLn1e4_64);
  float ang = (float)s * inv_freq;
  float sn, cs;
  __sincosf(ang, &sn, &cs);

  float q0 = (float)q[base + d], q1 = (float)q[base + d + 64];
  q[base + d]      = (__bf16)(q0 * cs - q1 * sn);
  q[base + d + 64] = (__bf16)(q1 * cs + q0 * sn);
  float k0 = (float)k[base + d], k1 = (float)k[base + d + 64];
  k[base + d]      = (__bf16)(k0 * cs - k1 * sn);
  k[base + d + 64] = (__bf16)(k1 * cs + k0 * sn);
}

// ---- Kernel 3: flash attention ---------------------------------------------
// One block = (b, h, 128 query rows); 8 waves, each owns 16 query rows.
// K block staged by the Tensor Data Mover (with LDS pad feature reproducing
// the [64][136] anti-bank-conflict layout); V staged via VGPRs (transpose).

__global__ __launch_bounds__(256)
void flash_attn_kernel(const __bf16* __restrict__ qg,
                       const __bf16* __restrict__ kg,
                       const __bf16* __restrict__ vg,
                       __bf16* __restrict__ ctxg) {
  __shared__ __bf16 ldk [64][136];     // K block  [key][hd], padded (64+4 DW rows)
  __shared__ __bf16 ldvT[128][72];     // V block transposed [hd][key], padded
  __shared__ __bf16 ldp [8][16][40];   // per-wave P tile [qrow][key], padded

  const int tid  = threadIdx.x;
  const int lane = tid & 31, w = tid >> 5;
  const int half = lane >> 4, lr = lane & 15;
  const int bh = blockIdx.y;
  const int b  = bh >> 4, h = bh & (H_ - 1);
  const int q0 = blockIdx.x * 128 + w * 16;
  const float scale = 0.08838834764831845f;  // 1/sqrt(128)

  // Q fragments (16 rows x 128 hd), kept in registers for whole kernel
  bf16x16 qf[4];
  {
    const __bf16* qrow = qg + (((size_t)b * S_ + q0 + lr) * H_ + h) * HD_;
#pragma unroll
    for (int c = 0; c < 4; ++c) qf[c] = a_frag(qrow + 32 * c, half);
  }

  v8f zero = {0.f, 0.f, 0.f, 0.f, 0.f, 0.f, 0.f, 0.f};
  v8f o[8];
#pragma unroll
  for (int f = 0; f < 8; ++f) o[f] = zero;
  float mi[8], li[8];
#pragma unroll
  for (int r = 0; r < 8; ++r) { mi[r] = -1e30f; li[r] = 0.f; }

  const int krow = tid & 63, kpart = tid >> 6;  // staging map: 64 keys x 4 hd-chunks

  for (int kb = 0; kb < S_; kb += 64) {
    const size_t rbase = (((size_t)b * S_ + kb + krow) * H_ + h) * HD_ + kpart * 32;

#if HAVE_TDM
    // ---- K block via Tensor Data Mover (wave 0 issues; uniform descriptor) --
    if (w == 0) {
      const unsigned lds_base = (unsigned)(uintptr_t)&ldk[0][0];  // LDS byte offset
      const unsigned long long ga =
          (unsigned long long)(uintptr_t)(kg + (((size_t)b * S_ + kb) * H_ + h) * HD_);
      // D# group0: count=1, lds_addr, global_addr[56:0], type=2
      u32x4 g0;
      g0[0] = 1u;
      g0[1] = lds_base;
      g0[2] = (unsigned)ga;
      g0[3] = (unsigned)((ga >> 32) & 0x01ffffffu) | (2u << 30);
      // D# group1: data_size=2B, pad_enable, pad_interval=64DW, pad_amount=4DW,
      // tensor_dim0=128, tensor_dim1=1<<20, tile=128x64, dim0_stride=2048
      i32x8 g1;
      g1[0] = (1 << 16) | (1 << 20) | (5 << 22) | (3 << 25);
      g1[1] = (int)(128u << 16);            // tensor_dim0[15:0] in bits[31:16]
      g1[2] = 0;                            // dim0 hi / dim1 lo16 = 0
      g1[3] = (int)(16u | (128u << 16));    // dim1[31:16]=16 (dim1=1M) | tile_dim0=128
      g1[4] = 64;                           // tile_dim1 = 64 keys
      g1[5] = 2048;                         // tensor_dim0_stride[31:0]
      g1[6] = 0;
      g1[7] = 0;
      i32x4 gz4 = {0, 0, 0, 0};
      i32x8 gz8 = {0, 0, 0, 0, 0, 0, 0, 0};
      // 6-arg form (clang-23 / therock-10.0 headers): groups 0..3 + extra + cpol
      __builtin_amdgcn_tensor_load_to_lds(g0, g1, gz4, gz4, gz8, 0);
    }
#endif

    // ---- V block transposed via VGPRs (all threads); K too on fallback ------
    {
      const __bf16* vsrc = vg + rbase;
#if !HAVE_TDM
      const __bf16* ksrc = kg + rbase;
#endif
#pragma unroll
      for (int e = 0; e < 4; ++e) {
#if !HAVE_TDM
        bf16x8 kv = *(const bf16x8*)(ksrc + e * 8);
        *(bf16x8*)(&ldk[krow][kpart * 32 + e * 8]) = kv;
#endif
        bf16x8 vv = *(const bf16x8*)(vsrc + e * 8);
#pragma unroll
        for (int q_ = 0; q_ < 8; ++q_) ldvT[kpart * 32 + e * 8 + q_][krow] = vv[q_];
      }
      if (kb + 64 < S_) {
        __builtin_prefetch(vg + rbase + (size_t)64 * H_ * HD_, 0, 1);
#if !HAVE_TDM
        __builtin_prefetch(kg + rbase + (size_t)64 * H_ * HD_, 0, 1);
#endif
      }
    }

#if HAVE_TDM
    if (w == 0) __builtin_amdgcn_s_wait_tensorcnt(0);  // K DMA complete
#endif
    __syncthreads();

#pragma unroll
    for (int cc = 0; cc < 2; ++cc) {
      const int kc = cc * 32;

      // Scores: two 16x16 tiles over 32 keys, K-dim = HD via 4 WMMAs each
      v8f sc[2]; sc[0] = zero; sc[1] = zero;
#pragma unroll
      for (int t = 0; t < 2; ++t) {
        const __bf16* kbase = &ldk[kc + t * 16 + lr][0];
#pragma unroll
        for (int c = 0; c < 4; ++c) {
          bf16x16 kf = b_frag(kbase + 32 * c, half);
          sc[t] = wmma_bf16(qf[c], kf, sc[t]);
        }
      }

      // Online softmax per row (row M = r + 8*half, reduce over 16 lanes)
#pragma unroll
      for (int r = 0; r < 8; ++r) {
        float s0 = sc[0][r] * scale;
        float s1 = sc[1][r] * scale;
        float rm = fmaxf(s0, s1);
        rm = fmaxf(rm, __shfl_xor(rm, 1));
        rm = fmaxf(rm, __shfl_xor(rm, 2));
        rm = fmaxf(rm, __shfl_xor(rm, 4));
        rm = fmaxf(rm, __shfl_xor(rm, 8));
        float mnew  = fmaxf(mi[r], rm);
        float alpha = __expf(mi[r] - mnew);
        float p0 = __expf(s0 - mnew);
        float p1 = __expf(s1 - mnew);
        float rs = p0 + p1;
        rs += __shfl_xor(rs, 1);
        rs += __shfl_xor(rs, 2);
        rs += __shfl_xor(rs, 4);
        rs += __shfl_xor(rs, 8);
        li[r] = li[r] * alpha + rs;
        mi[r] = mnew;
        ldp[w][8 * half + r][lr]      = (__bf16)p0;
        ldp[w][8 * half + r][16 + lr] = (__bf16)p1;
#pragma unroll
        for (int f = 0; f < 8; ++f) o[f][r] = o[f][r] * alpha;
      }
      // P went through per-wave LDS; wait for DS stores before re-reading
      asm volatile("s_wait_dscnt 0" ::: "memory");

      // ctx += P(16x32) * V(32 x 128)
      bf16x16 pf = a_frag(&ldp[w][lr][0], half);
#pragma unroll
      for (int f = 0; f < 8; ++f) {
        bf16x16 vf = b_frag(&ldvT[f * 16 + lr][kc], half);
        o[f] = wmma_bf16(pf, vf, o[f]);
      }
    }
    __syncthreads();
  }

  // Normalize and store ctx (bf16) at (b, qrow, h, hd)
#pragma unroll
  for (int r = 0; r < 8; ++r) {
    float inv = 1.0f / li[r];
    const size_t row = ((size_t)b * S_ + q0 + 8 * half + r) * H_ + h;
#pragma unroll
    for (int f = 0; f < 8; ++f)
      ctxg[row * HD_ + f * 16 + lr] = (__bf16)(o[f][r] * inv);
  }
}

// ---- Kernel 4: output projection GEMM (bf16 A, fp32 B -> fp32 out) ---------

__global__ __launch_bounds__(256)
void out_gemm_kernel(const __bf16* __restrict__ ctx,
                     const float* __restrict__ wo,
                     float* __restrict__ out) {
  constexpr int K = H_ * HD_, N = D_;

  __shared__ __bf16 lda[128][40];
  __shared__ __bf16 ldb[128][40];

  const int tid  = threadIdx.x;
  const int lane = tid & 31, w = tid >> 5;
  const int half = lane >> 4, lr = lane & 15;
  const int wm = (w & 1) * 64;
  const int wn = (w >> 1) * 32;
  const int m0 = blockIdx.y * 128;
  const int n0 = blockIdx.x * 128;

  v8f zero = {0.f, 0.f, 0.f, 0.f, 0.f, 0.f, 0.f, 0.f};
  v8f acc[4][2];
#pragma unroll
  for (int i = 0; i < 4; ++i)
#pragma unroll
    for (int j = 0; j < 2; ++j) acc[i][j] = zero;

  const int arow = tid & 127, ahalf = tid >> 7;
  const int brow = tid & 31,  bpart = tid >> 5;

  for (int kb = 0; kb < K; kb += 32) {
    {
      const __bf16* src = ctx + (size_t)(m0 + arow) * K + kb + ahalf * 16;
      *(bf16x8*)(&lda[arow][ahalf * 16 + 0]) = *(const bf16x8*)(src + 0);
      *(bf16x8*)(&lda[arow][ahalf * 16 + 8]) = *(const bf16x8*)(src + 8);
    }
    {
      const float* src = wo + (size_t)(kb + brow) * N + n0 + bpart * 16;
      f32x4 v0 = *(const f32x4*)(src + 0);
      f32x4 v1 = *(const f32x4*)(src + 4);
      f32x4 v2 = *(const f32x4*)(src + 8);
      f32x4 v3 = *(const f32x4*)(src + 12);
#pragma unroll
      for (int e = 0; e < 4; ++e) {
        ldb[bpart * 16 + e +  0][brow] = (__bf16)v0[e];
        ldb[bpart * 16 + e +  4][brow] = (__bf16)v1[e];
        ldb[bpart * 16 + e +  8][brow] = (__bf16)v2[e];
        ldb[bpart * 16 + e + 12][brow] = (__bf16)v3[e];
      }
    }
    if (kb + 32 < K) {
      __builtin_prefetch(ctx + (size_t)(m0 + arow) * K + kb + 32 + ahalf * 16, 0, 1);
      __builtin_prefetch(wo + (size_t)(kb + 32 + brow) * N + n0 + bpart * 16, 0, 1);
    }
    __syncthreads();

    bf16x16 af[4], bfr[2];
#pragma unroll
    for (int i = 0; i < 4; ++i) af[i] = a_frag(&lda[wm + i * 16 + lr][0], half);
#pragma unroll
    for (int j = 0; j < 2; ++j) bfr[j] = b_frag(&ldb[wn + j * 16 + lr][0], half);
#pragma unroll
    for (int i = 0; i < 4; ++i)
#pragma unroll
      for (int j = 0; j < 2; ++j) acc[i][j] = wmma_bf16(af[i], bfr[j], acc[i][j]);
    __syncthreads();
  }

#pragma unroll
  for (int i = 0; i < 4; ++i) {
    const int rowb = m0 + wm + i * 16 + 8 * half;
#pragma unroll
    for (int j = 0; j < 2; ++j) {
      const int col = n0 + wn + j * 16 + lr;
#pragma unroll
      for (int r = 0; r < 8; ++r)
        out[(size_t)(rowb + r) * N + col] = acc[i][j][r];
    }
  }
}

// ---- Launch ----------------------------------------------------------------

extern "C" void kernel_launch(void* const* d_in, const int* in_sizes, int n_in,
                              void* d_out, int out_size, void* d_ws, size_t ws_size,
                              hipStream_t stream) {
  (void)in_sizes; (void)n_in; (void)out_size; (void)ws_size;
  const float* x  = (const float*)d_in[0];
  const float* wq = (const float*)d_in[1];
  const float* wk = (const float*)d_in[2];
  const float* wv = (const float*)d_in[3];
  const float* wo = (const float*)d_in[4];
  float* out = (float*)d_out;

  const size_t elems = (size_t)B_ * S_ * H_ * HD_;  // 16 Mi per tensor
  __bf16* qws = (__bf16*)d_ws;
  __bf16* kws = qws + elems;
  __bf16* vws = kws + elems;
  __bf16* cws = vws + elems;

  dim3 g1(D_ / 128, (B_ * S_) / 128, 3);
  qkv_gemm_kernel<<<g1, 256, 0, stream>>>(x, wq, wk, wv, qws, kws, vws);

  dim3 g2(B_ * S_, H_ / 4);
  rope_kernel<<<g2, 256, 0, stream>>>(qws, kws);

  dim3 g3(S_ / 128, B_ * H_);
  flash_attn_kernel<<<g3, 256, 0, stream>>>(qws, kws, vws, cws);

  dim3 g4(D_ / 128, (B_ * S_) / 128);
  out_gemm_kernel<<<g4, 256, 0, stream>>>(cws, wo, out);
}
